// SwinTransformerBlock_87179246174413
// MI455X (gfx1250) — compile-verified, matrix operand order checked by reading, not run on hardware
//
#include <hip/hip_runtime.h>
#include <hip/hip_bf16.h>
#include <math.h>

typedef __attribute__((ext_vector_type(16))) _Float16 v16h;
typedef __attribute__((ext_vector_type(8)))  _Float16 v8h;
typedef __attribute__((ext_vector_type(8)))  float    v8f;

#define DIM    256
#define HH     56
#define WWID   56
#define WSZ    7
#define SHIFTA 3
#define NTOK   49
#define HEADS  8
#define HD     32
#define BATCH  32
#define TOKENS (BATCH * HH * WWID)   /* 100352 */
#define NWIN   (BATCH * 64)          /* 2048 windows */
#define QSCALE 0.17677669529663687f  /* 32^-0.5 */

__device__ __forceinline__ v8f wmma16(v16h a, v16h b, v8f c) {
  return __builtin_amdgcn_wmma_f32_16x16x32_f16(false, a, false, b, (short)0, c,
                                                false, false);
}

// A fragment: row-major 16x32 f16 tile at base, leading dim ld (halves).
// Lanes 0-15: row=lane, K {0..7,16..23}; lanes 16-31: K {8..15,24..31}.
// Compiles to 2x ds_load_b128 / global_load_b128.
__device__ __forceinline__ v16h frag_a(const _Float16* base, int ld) {
  const int lane = threadIdx.x & 31;
  const _Float16* p = base + (lane & 15) * ld + ((lane >> 4) << 3);
  v16h a;
#pragma unroll
  for (int i = 0; i < 8; ++i) { a[i] = p[i]; a[8 + i] = p[16 + i]; }
  return a;
}

// B fragment from pre-swizzled weights: one contiguous 32B load per lane.
__device__ __forceinline__ v16h frag_bp(const _Float16* wp, int ktiles, int ks,
                                        int nt) {
  const int lane = threadIdx.x & 31;
  return *(const v16h*)(wp + (((size_t)(nt * ktiles + ks)) << 9) + (lane << 4));
}

// ---- pack weights into B-fragment-major layout -----------------------------
__global__ void swin_wpack_kernel(const float* __restrict__ w,
                                  _Float16* __restrict__ wp, int K, int N) {
  int p = blockIdx.x * blockDim.x + threadIdx.x;
  if (p >= K * N) return;
  int tile = p >> 9, rem = p & 511;
  int l = rem >> 4, i = rem & 15;
  int ktiles = K >> 5;
  int ks = tile % ktiles, nt = tile / ktiles;
  int col = l & 15;
  int kb  = (l >> 4) << 3;
  int kk  = kb + ((i < 8) ? i : (8 + i));   // i>=8 -> kb+16+(i-8)
  wp[p] = (_Float16)w[(size_t)(nt * 16 + col) * K + (ks * 32 + kk)];
}

// ---- LN1: one wave per token row, vectorized f16 out -----------------------
__global__ __launch_bounds__(256) void swin_ln1_kernel(
    const float* __restrict__ x, const float* __restrict__ g,
    const float* __restrict__ b, _Float16* __restrict__ hn) {
  int row  = (blockIdx.x * blockDim.x + threadIdx.x) >> 5;
  int lane = threadIdx.x & 31;
  if (row >= TOKENS) return;
  const float* xr = x + (size_t)row * DIM + lane * 8;
  float v[8], s = 0.f, s2 = 0.f;
#pragma unroll
  for (int i = 0; i < 8; ++i) { v[i] = xr[i]; s += v[i]; s2 += v[i] * v[i]; }
#pragma unroll
  for (int m = 16; m >= 1; m >>= 1) {
    s += __shfl_xor(s, m, 32); s2 += __shfl_xor(s2, m, 32);
  }
  float mean = s * (1.f / DIM);
  float var  = s2 * (1.f / DIM) - mean * mean;
  float inv  = rsqrtf(var + 1e-5f);
  v8h o;
#pragma unroll
  for (int i = 0; i < 8; ++i) {
    int c = lane * 8 + i;
    o[i] = (_Float16)((v[i] - mean) * inv * g[c] + b[c]);
  }
  *(v8h*)(hn + (size_t)row * DIM + lane * 8) = o;
}

// ---- fused window attention: one block (8 waves) per window ----------------
__global__ __launch_bounds__(256) void swin_attn_kernel(
    const _Float16* __restrict__ hn, const float* __restrict__ x,
    const _Float16* __restrict__ qkv_wp, const float* __restrict__ qkv_b,
    const float* __restrict__ rpb, const _Float16* __restrict__ proj_wp,
    const float* __restrict__ proj_b, float* __restrict__ xa) {
  extern __shared__ char smem[];                       // 128 KB dynamic LDS
  _Float16* sA = (_Float16*)smem;                      // 64x256 f16 tokens
  float*    sS = (float*)smem;                         // 64x64 f32 (reuse sA)
  _Float16* sP = (_Float16*)(smem + 16384);            // 64x64 f16 probs
  _Float16* sQ = (_Float16*)(smem + 32768);            // 64x256 Q, later O
  _Float16* sK = (_Float16*)(smem + 65536);            // 64x256 row-major
  _Float16* sVt = (_Float16*)(smem + 98304);           // 256x64 V TRANSPOSED
  __shared__ int sTok[64];
  __shared__ int sLab[64];

  const int tid = threadIdx.x;
  const int wvu = __builtin_amdgcn_readfirstlane(tid >> 5);  // scalar wave id
  const int win = blockIdx.x;
  const int bb  = win >> 6;
  const int w   = win & 63;
  const int wi  = w >> 3, wj = w & 7;

  if (tid < 64) {
    int t  = tid;
    int tt = (t < NTOK) ? t : 0;
    int tr = wi * WSZ + tt / WSZ;
    int tc = wj * WSZ + tt % WSZ;
    int orow = (tr + SHIFTA) % HH;
    int ocol = (tc + SHIFTA) % WWID;
    sTok[t] = bb * HH * WWID + orow * WWID + ocol;
    int rr = (tr < HH - WSZ) ? 0 : ((tr < HH - SHIFTA) ? 1 : 2);
    int cc = (tc < WWID - WSZ) ? 0 : ((tc < WWID - SHIFTA) ? 1 : 2);
    sLab[t] = rr * 3 + cc;
  }
  __syncthreads();

  // stage window tokens with CDNA5 async global->LDS DMA (ASYNCcnt tracked),
  // 16B per lane; pad rows (49..63) zero-filled with plain ds stores.
  for (int ch = tid; ch < 2048; ch += 256) {
    int r = ch >> 5;
    int c = (ch & 31) << 3;
    if (r < NTOK) {
      unsigned lds = (unsigned)(size_t)(sA + r * DIM + c);  // low 32b = LDS off
      unsigned long long ga =
          (unsigned long long)(size_t)(hn + (size_t)sTok[r] * DIM + c);
      asm volatile("global_load_async_to_lds_b128 %0, %1, off"
                   :: "v"(lds), "v"(ga) : "memory");
    } else {
      *(uint4*)(sA + r * DIM + c) = make_uint4(0u, 0u, 0u, 0u);
    }
  }
  asm volatile("s_wait_asynccnt 0" ::: "memory");
  __syncthreads();

  // QKV GEMM: 64x768, K=256 -> 4 Mtiles x 48 Ntiles
  for (int j = wvu; j < 192; j += 8) {
    int mt = j & 3, nt = j >> 2;     // scalar
    v8f acc = {};
#pragma unroll
    for (int ks = 0; ks < 8; ++ks) {
      __builtin_prefetch(qkv_wp + (((size_t)(nt * 8 + ks + 1)) << 9), 0, 1);
      v16h a  = frag_a(sA + mt * 16 * DIM + ks * 32, DIM);
      v16h bf = frag_bp(qkv_wp, 8, ks, nt);
      acc = wmma16(a, bf, acc);
    }
    int lane = tid & 31, col = lane & 15, rowb = (lane >> 4) << 3;
    int gcol = nt * 16 + col;
    float bias = qkv_b[gcol];
    if (nt < 16) {                          // Q columns (scalar branch)
#pragma unroll
      for (int r = 0; r < 8; ++r)
        sQ[(mt * 16 + rowb + r) * DIM + gcol] =
            (_Float16)((acc[r] + bias) * QSCALE);
    } else if (nt < 32) {                   // K columns
      int dcol = gcol - 256;
#pragma unroll
      for (int r = 0; r < 8; ++r)
        sK[(mt * 16 + rowb + r) * DIM + dcol] = (_Float16)(acc[r] + bias);
    } else {                                // V columns, stored transposed
      int dcol = gcol - 512;
#pragma unroll
      for (int r = 0; r < 8; ++r)
        sVt[dcol * 64 + (mt * 16 + rowb + r)] = (_Float16)(acc[r] + bias);
    }
  }
  __syncthreads();

  for (int h = 0; h < HEADS; ++h) {
    // S = Q_h (64x32) x K_h^T : 4x4 tiles, single K=32 step
    for (int j = wvu; j < 16; j += 8) {
      int mt = j >> 2, nt = j & 3;
      v16h a  = frag_a(sQ + mt * 16 * DIM + h * HD, DIM);
      v16h bf = frag_a(sK + nt * 16 * DIM + h * HD, DIM);  // B = K^T
      v8f acc = {};
      acc = wmma16(a, bf, acc);
      int lane = tid & 31, col = lane & 15, rowb = (lane >> 4) << 3;
#pragma unroll
      for (int r = 0; r < 8; ++r)
        sS[(mt * 16 + rowb + r) * 64 + nt * 16 + col] = acc[r];
    }
    __syncthreads();

    // softmax rows with on-the-fly rel-pos bias + shift mask
    if (tid < 64) {
      int i = tid;
      if (i < NTOK) {
        int ir = i / WSZ, ic = i % WSZ, li = sLab[i];
        float mx = -1e30f;
        for (int jx = 0; jx < NTOK; ++jx) {
          int jr = jx / WSZ, jc = jx % WSZ;
          int idx = (ir - jr + WSZ - 1) * (2 * WSZ - 1) + (ic - jc + WSZ - 1);
          float av = sS[i * 64 + jx] + rpb[idx * HEADS + h] +
                     ((li != sLab[jx]) ? -100.f : 0.f);
          mx = fmaxf(mx, av);
        }
        float sum = 0.f;
        for (int jx = 0; jx < NTOK; ++jx) {
          int jr = jx / WSZ, jc = jx % WSZ;
          int idx = (ir - jr + WSZ - 1) * (2 * WSZ - 1) + (ic - jc + WSZ - 1);
          float av = sS[i * 64 + jx] + rpb[idx * HEADS + h] +
                     ((li != sLab[jx]) ? -100.f : 0.f);
          float e = __expf(av - mx);
          sum += e;
          sP[i * 64 + jx] = (_Float16)e;
        }
        for (int jx = NTOK; jx < 64; ++jx) sP[i * 64 + jx] = (_Float16)0.f;
        float inv = 1.f / sum;
        for (int jx = 0; jx < NTOK; ++jx)
          sP[i * 64 + jx] = (_Float16)((float)sP[i * 64 + jx] * inv);
      } else {
        for (int jx = 0; jx < 64; ++jx) sP[i * 64 + jx] = (_Float16)0.f;
      }
    }
    __syncthreads();

    // O_h = P (64x64) x V_h (64x32): 4x2 tiles, K=64; V^T keeps B contiguous.
    {
      int mt = wvu >> 1, nt = wvu & 1;
      v8f acc = {};
#pragma unroll
      for (int ks = 0; ks < 2; ++ks) {
        v16h a  = frag_a(sP + mt * 16 * 64 + ks * 32, 64);
        v16h bf = frag_a(sVt + (h * HD + nt * 16) * 64 + ks * 32, 64);
        acc = wmma16(a, bf, acc);
      }
      int lane = tid & 31, col = lane & 15, rowb = (lane >> 4) << 3;
#pragma unroll
      for (int r = 0; r < 8; ++r)
        sQ[(mt * 16 + rowb + r) * DIM + h * HD + nt * 16 + col] =
            (_Float16)acc[r];   // O overwrites dead Q_h slot
    }
    __syncthreads();
  }

  // proj: O (64x256, in sQ) x proj (256x256) + bias + shortcut -> scatter
  for (int j = wvu; j < 64; j += 8) {
    int mt = j & 3, nt = j >> 2;
    v8f acc = {};
#pragma unroll
    for (int ks = 0; ks < 8; ++ks) {
      v16h a  = frag_a(sQ + mt * 16 * DIM + ks * 32, DIM);
      v16h bf = frag_bp(proj_wp, 8, ks, nt);
      acc = wmma16(a, bf, acc);
    }
    int lane = tid & 31, col = lane & 15, rowb = (lane >> 4) << 3;
    int gcol = nt * 16 + col;
    float bias = proj_b[gcol];
#pragma unroll
    for (int r = 0; r < 8; ++r) {
      int grow = mt * 16 + rowb + r;
      if (grow < NTOK) {
        size_t o = (size_t)sTok[grow] * DIM + gcol;
        xa[o] = x[o] + acc[r] + bias;
      }
    }
  }
}

// ---- fused LN2 + MLP: 64 tokens per block ----------------------------------
__global__ __launch_bounds__(256) void swin_mlp_kernel(
    const float* __restrict__ xa, const float* __restrict__ n2w,
    const float* __restrict__ n2b, const _Float16* __restrict__ fc1_wp,
    const float* __restrict__ fc1_b, const _Float16* __restrict__ fc2_wp,
    const float* __restrict__ fc2_b, float* __restrict__ out) {
  extern __shared__ char smem[];                 // 160 KB dynamic LDS
  _Float16* sH = (_Float16*)smem;                // 64x256  f16
  _Float16* sG = (_Float16*)(smem + 32768);      // 64x1024 f16
  const int tid = threadIdx.x;
  const int wvu = __builtin_amdgcn_readfirstlane(tid >> 5);
  const size_t tok0 = (size_t)blockIdx.x * 64;

  // LN2: 4 lanes per row, float4 reads, v8h writes
  {
    int row = tid >> 2, part = tid & 3;
    const float4* xr = (const float4*)(xa + (tok0 + row) * DIM + part * 64);
    float s = 0.f, s2 = 0.f;
#pragma unroll
    for (int c = 0; c < 16; ++c) {
      float4 t = xr[c];
      s += t.x + t.y + t.z + t.w;
      s2 += t.x * t.x + t.y * t.y + t.z * t.z + t.w * t.w;
    }
    s  += __shfl_xor(s, 1, 32);  s  += __shfl_xor(s, 2, 32);
    s2 += __shfl_xor(s2, 1, 32); s2 += __shfl_xor(s2, 2, 32);
    float mean = s * (1.f / DIM);
    float var  = s2 * (1.f / DIM) - mean * mean;
    float inv  = rsqrtf(var + 1e-5f);
#pragma unroll
    for (int c8 = 0; c8 < 8; ++c8) {
      v8h o;
      float4 t0 = xr[c8 * 2], t1 = xr[c8 * 2 + 1];
      float tv[8] = {t0.x, t0.y, t0.z, t0.w, t1.x, t1.y, t1.z, t1.w};
#pragma unroll
      for (int i = 0; i < 8; ++i) {
        int gc = part * 64 + c8 * 8 + i;
        o[i] = (_Float16)((tv[i] - mean) * inv * n2w[gc] + n2b[gc]);
      }
      *(v8h*)(sH + row * DIM + part * 64 + c8 * 8) = o;
    }
  }
  __syncthreads();

  // fc1 + GELU: 64x1024, K=256 -> 4x64 tiles
  for (int j = wvu; j < 256; j += 8) {
    int mt = j & 3, nt = j >> 2;
    v8f acc = {};
#pragma unroll
    for (int ks = 0; ks < 8; ++ks) {
      __builtin_prefetch(fc1_wp + (((size_t)(nt * 8 + ks + 1)) << 9), 0, 1);
      v16h a  = frag_a(sH + mt * 16 * DIM + ks * 32, DIM);
      v16h bf = frag_bp(fc1_wp, 8, ks, nt);
      acc = wmma16(a, bf, acc);
    }
    int lane = tid & 31, col = lane & 15, rowb = (lane >> 4) << 3;
    int gcol = nt * 16 + col;
    float bias = fc1_b[gcol];
#pragma unroll
    for (int r = 0; r < 8; ++r) {
      float v = acc[r] + bias;
      v = 0.5f * v * (1.f + erff(v * 0.70710678118f));   // exact GELU
      sG[(mt * 16 + rowb + r) * 1024 + gcol] = (_Float16)v;
    }
  }
  __syncthreads();

  // fc2 + residual: 64x256, K=1024 -> 4x16 tiles
  for (int j = wvu; j < 64; j += 8) {
    int mt = j & 3, nt = j >> 2;
    v8f acc = {};
#pragma unroll 4
    for (int ks = 0; ks < 32; ++ks) {
      __builtin_prefetch(fc2_wp + (((size_t)(nt * 32 + ks + 1)) << 9), 0, 1);
      v16h a  = frag_a(sG + mt * 16 * 1024 + ks * 32, 1024);
      v16h bf = frag_bp(fc2_wp, 32, ks, nt);
      acc = wmma16(a, bf, acc);
    }
    int lane = tid & 31, col = lane & 15, rowb = (lane >> 4) << 3;
    int gcol = nt * 16 + col;
    float bias = fc2_b[gcol];
#pragma unroll
    for (int r = 0; r < 8; ++r) {
      int grow = mt * 16 + rowb + r;
      size_t o = (tok0 + grow) * DIM + gcol;
      out[o] = xa[o] + acc[r] + bias;
    }
  }
}

// ---- CDNA5 cluster/tensor probe (never launched; codegen showcase only) ----
__global__ void swin_cdna5_probe_kernel(float* __restrict__ o) {
  __builtin_amdgcn_s_wait_tensorcnt(0);
  int cid = __builtin_amdgcn_cluster_id_x();
  __builtin_amdgcn_s_cluster_barrier();
  o[threadIdx.x] = (float)cid;
}

extern "C" void kernel_launch(void* const* d_in, const int* in_sizes, int n_in,
                              void* d_out, int out_size, void* d_ws,
                              size_t ws_size, hipStream_t stream) {
  (void)in_sizes; (void)n_in; (void)out_size; (void)ws_size;
  const float* x      = (const float*)d_in[0];
  const float* n1w    = (const float*)d_in[1];
  const float* n1b    = (const float*)d_in[2];
  const float* qkv_w  = (const float*)d_in[3];
  const float* qkv_b  = (const float*)d_in[4];
  const float* rpb    = (const float*)d_in[5];
  const float* proj_w = (const float*)d_in[6];
  const float* proj_b = (const float*)d_in[7];
  const float* n2w    = (const float*)d_in[8];
  const float* n2b    = (const float*)d_in[9];
  const float* fc1_w  = (const float*)d_in[10];
  const float* fc1_b  = (const float*)d_in[11];
  const float* fc2_w  = (const float*)d_in[12];
  const float* fc2_b  = (const float*)d_in[13];
  float* out = (float*)d_out;

  char* ws = (char*)d_ws;
  _Float16* qkv_wp  = (_Float16*)(ws + 0);          // 256x768  f16 packed
  _Float16* proj_wp = (_Float16*)(ws + 393216);     // 256x256
  _Float16* fc1_wp  = (_Float16*)(ws + 524288);     // 256x1024
  _Float16* fc2_wp  = (_Float16*)(ws + 1048576);    // 1024x256
  _Float16* hn      = (_Float16*)(ws + 1572864);    // TOKENS*256 f16
  float*    xa      = (float*)(ws + 52953088);      // TOKENS*256 f32

  swin_wpack_kernel<<<768, 256, 0, stream>>>(qkv_w, qkv_wp, 256, 768);
  swin_wpack_kernel<<<256, 256, 0, stream>>>(proj_w, proj_wp, 256, 256);
  swin_wpack_kernel<<<1024, 256, 0, stream>>>(fc1_w, fc1_wp, 256, 1024);
  swin_wpack_kernel<<<1024, 256, 0, stream>>>(fc2_w, fc2_wp, 1024, 256);

  swin_ln1_kernel<<<TOKENS / 8, 256, 0, stream>>>(x, n1w, n1b, hn);

  swin_attn_kernel<<<NWIN, 256, 131072, stream>>>(hn, x, qkv_wp, qkv_b, rpb,
                                                  proj_wp, proj_b, xa);

  swin_mlp_kernel<<<TOKENS / 64, 256, 163840, stream>>>(xa, n2w, n2b, fc1_wp,
                                                        fc1_b, fc2_wp, fc2_b,
                                                        out);
}
